// HeteroGNN_52913997087089
// MI455X (gfx1250) — compile-verified, matrix operand order checked by reading, not run on hardware
//
#include <hip/hip_runtime.h>
#include <hip/hip_bf16.h>
#include <math.h>

#define D 128
#define N_TEAM 50000
#define N_PLAYER 400000
#define N_GAMES 100000

typedef __bf16 bf16_t;
typedef __attribute__((ext_vector_type(16))) __bf16 v16bf;
typedef __attribute__((ext_vector_type(8)))  float  v8f;

__device__ inline bf16_t f2bf(float f) {
    unsigned u = __float_as_uint(f);
    unsigned r = (u + 0x7FFFu + ((u >> 16) & 1u)) >> 16;  // round-to-nearest-even
    unsigned short s = (unsigned short)r;
    return *reinterpret_cast<bf16_t*>(&s);
}

// pack bf16(f0) into [15:0] and bf16(f1) into [31:16] with one v_perm_b32
__device__ inline unsigned pk2bf(float f0, float f1) {
    unsigned u0 = __float_as_uint(f0) + 0x8000u;  // round-half-up
    unsigned u1 = __float_as_uint(f1) + 0x8000u;
    return __builtin_amdgcn_perm(u1, u0, 0x07060302u);
}

__device__ inline float leaky02(float x) { return x > 0.f ? x : 0.2f * x; }

__device__ inline void atomicMaxF(float* addr, float val) {
    int* ai = (int*)addr;
    int old = __float_as_int(*addr);
    while (__int_as_float(old) < val) {
        int prev = atomicCAS(ai, old, __float_as_int(val));
        if (prev == old) break;
        old = prev;
    }
}

// ---------------- elementwise / fill ----------------

__global__ void k_fill(float* __restrict__ p, long n, float v) {
    long i = (long)blockIdx.x * blockDim.x + threadIdx.x;
    if (i < n) p[i] = v;
}

__global__ void k_rsqrt(float* __restrict__ p, int n) {
    int i = blockIdx.x * blockDim.x + threadIdx.x;
    if (i < n) p[i] = rsqrtf(p[i]);
}

__global__ void k_relu(float* __restrict__ p, long n) {
    long i = (long)blockIdx.x * blockDim.x + threadIdx.x;
    if (i < n) p[i] = fmaxf(p[i], 0.f);
}

// out[i][c] = sum of up to 6 bias vectors (null -> skip)
__global__ void k_init_bias(float* __restrict__ out, long N,
                            const float* b0, const float* b1, const float* b2,
                            const float* b3, const float* b4, const float* b5) {
    long i = (long)blockIdx.x * blockDim.x + threadIdx.x;
    if (i >= N * D) return;
    int c = (int)(i & (D - 1));
    float v = 0.f;
    if (b0) v += b0[c];
    if (b1) v += b1[c];
    if (b2) v += b2[c];
    if (b3) v += b3[c];
    if (b4) v += b4[c];
    if (b5) v += b5[c];
    out[i] = v;
}

// ---------------- gathers ----------------

__global__ void k_gather_emb(const float* __restrict__ emb, const int* __restrict__ ids,
                             float* __restrict__ out, int N) {
    long i = (long)blockIdx.x * blockDim.x + threadIdx.x;  // one float4 each
    if (i >= (long)N * 32) return;
    int row = (int)(i >> 5), q = (int)(i & 31);
    ((float4*)out)[i] = ((const float4*)emb)[(long)ids[row] * 32 + q];
}

// ---------------- GCN pieces ----------------

__global__ void k_edge_deg(const int* __restrict__ dst, int E, float* __restrict__ deg) {
    int i = blockIdx.x * blockDim.x + threadIdx.x;
    if (i < E) atomicAdd(&deg[dst[i]], 1.0f);
}

// H[i][c] = X[i][c] * dis[i]^2   (self loop contribution; also zero-inits rest)
__global__ void k_selfloop(const float* __restrict__ X, const float* __restrict__ dis,
                           float* __restrict__ H, int N) {
    long i = (long)blockIdx.x * blockDim.x + threadIdx.x;
    if (i >= (long)N * D) return;
    int node = (int)(i >> 7);
    float w = dis[node];
    H[i] = X[i] * w * w;
}

// wave per edge: H[dst] += X[src] * dis[src]*dis[dst]
__global__ void k_gcn_scatter(const float* __restrict__ X, const int* __restrict__ src,
                              const int* __restrict__ dst, int E,
                              const float* __restrict__ dis, float* __restrict__ H) {
    long t = (long)blockIdx.x * blockDim.x + threadIdx.x;
    int e = (int)(t >> 5);
    if (e >= E) return;
    int lane = threadIdx.x & 31;
    int s = src[e], d = dst[e];
    float w = dis[s] * dis[d];
    float4 x = ((const float4*)(X + (long)s * D))[lane];
    float* hp = H + (long)d * D + lane * 4;
    atomicAdd(hp + 0, x.x * w);
    atomicAdd(hp + 1, x.y * w);
    atomicAdd(hp + 2, x.z * w);
    atomicAdd(hp + 3, x.w * w);
}

// ---------------- GAT pieces ----------------

// wa[k] = sum_c W[k][c] * a[c]
__global__ void k_wa(const float* __restrict__ W, const float* __restrict__ a,
                     float* __restrict__ wa) {
    int k = threadIdx.x;
    float s = 0.f;
    for (int c = 0; c < D; ++c) s += W[k * D + c] * a[c];
    wa[k] = s;
}

// out[r] = dot(X[r][:], v[:]) — wave per row
__global__ void k_dot_rows(const float* __restrict__ X, const float* __restrict__ v,
                           float* __restrict__ out, int N) {
    long t = (long)blockIdx.x * blockDim.x + threadIdx.x;
    int r = (int)(t >> 5);
    if (r >= N) return;
    int lane = threadIdx.x & 31;
    float4 x = ((const float4*)(X + (long)r * D))[lane];
    float4 y = ((const float4*)v)[lane];
    float s = x.x * y.x + x.y * y.y + x.z * y.z + x.w * y.w;
    #pragma unroll
    for (int off = 16; off > 0; off >>= 1) s += __shfl_xor(s, off, 32);
    if (lane == 0) out[r] = s;
}

__global__ void k_gat_max(const int* __restrict__ src, const int* __restrict__ dst,
                          int E, int m, const float* __restrict__ als,
                          const float* __restrict__ ald, float* __restrict__ emax) {
    int i = blockIdx.x * blockDim.x + threadIdx.x;
    if (i >= E + m) return;
    int s, d;
    if (i < E) { s = src[i]; d = dst[i]; if (s == d) return; }
    else       { s = d = i - E; }
    atomicMaxF(&emax[d], leaky02(als[s] + ald[d]));
}

__global__ void k_gat_den(const int* __restrict__ src, const int* __restrict__ dst,
                          int E, int m, const float* __restrict__ als,
                          const float* __restrict__ ald, const float* __restrict__ emax,
                          float* __restrict__ den) {
    int i = blockIdx.x * blockDim.x + threadIdx.x;
    if (i >= E + m) return;
    int s, d;
    if (i < E) { s = src[i]; d = dst[i]; if (s == d) return; }
    else       { s = d = i - E; }
    float ev = leaky02(als[s] + ald[d]);
    atomicAdd(&den[d], expf(ev - emax[d]));
}

// wave per (edge or self-loop): H[dst] += alpha * Xs[src]
__global__ void k_gat_scatter(const float* __restrict__ Xs, const int* __restrict__ src,
                              const int* __restrict__ dst, int E, int m,
                              const float* __restrict__ als, const float* __restrict__ ald,
                              const float* __restrict__ emax, const float* __restrict__ den,
                              float* __restrict__ H) {
    long t = (long)blockIdx.x * blockDim.x + threadIdx.x;
    int e = (int)(t >> 5);
    if (e >= E + m) return;
    int lane = threadIdx.x & 31;
    int s, d;
    if (e < E) { s = src[e]; d = dst[e]; if (s == d) return; }
    else       { s = d = e - E; }
    float ev = leaky02(als[s] + ald[d]);
    float alpha = expf(ev - emax[d]) / fmaxf(den[d], 1e-16f);
    float4 x = ((const float4*)(Xs + (long)s * D))[lane];
    float* hp = H + (long)d * D + lane * 4;
    atomicAdd(hp + 0, x.x * alpha);
    atomicAdd(hp + 1, x.y * alpha);
    atomicAdd(hp + 2, x.z * alpha);
    atomicAdd(hp + 3, x.w * alpha);
}

// ---------------- weight pre-swizzle (f32 -> bf16 in WMMA B-operand order) ------
// out[((kt*8 + nt)*32 + lane)*16 + j] holds the 16 bf16 values lane `lane` feeds
// to WMMA for k-chunk kt, n-tile nt.  W is [KT*32, 128] row-major f32.
__global__ void k_wswz(const float* __restrict__ W, bf16_t* __restrict__ out, int KT) {
    int t = blockIdx.x * blockDim.x + threadIdx.x;  // one (kt, nt, lane)
    if (t >= KT * 8 * 32) return;
    int lane = t & 31;
    int nt   = (t >> 5) & 7;
    int kt   = t >> 8;
    int koff = (lane & 16) ? 8 : 0;
    int n    = nt * 16 + (lane & 15);
    int kb   = kt * 32 + koff;
    bf16_t* dst = out + (long)t * 16;
    #pragma unroll
    for (int j = 0; j < 8; ++j) {
        dst[j]     = f2bf(W[(kb + j) * D + n]);
        dst[8 + j] = f2bf(W[(kb + 16 + j) * D + n]);
    }
}

// build one 16-element A fragment (per-lane) from 4 float4s using v_perm packs
union afrag_u { v16bf v; unsigned w[8]; };
__device__ inline v16bf make_afrag(float4 p0, float4 p1, float4 p2, float4 p3) {
    afrag_u a;
    a.w[0] = pk2bf(p0.x, p0.y);
    a.w[1] = pk2bf(p0.z, p0.w);
    a.w[2] = pk2bf(p1.x, p1.y);
    a.w[3] = pk2bf(p1.z, p1.w);
    a.w[4] = pk2bf(p2.x, p2.y);
    a.w[5] = pk2bf(p2.z, p2.w);
    a.w[6] = pk2bf(p3.x, p3.y);
    a.w[7] = pk2bf(p3.z, p3.w);
    return a.v;
}

// ---------------- WMMA GEMMs (bf16 inputs, f32 accumulate) ----------------
// Out[N,128] += X[N,128] @ W; W pre-swizzled to bf16 fragments.
// One wave computes 16 rows x all 128 cols; existing Out preloaded as C.
__global__ void k_gemm_acc(const float* __restrict__ X, const bf16_t* __restrict__ Wswz,
                           float* __restrict__ Out, int nTiles) {
    long t = (long)blockIdx.x * blockDim.x + threadIdx.x;
    int wave = (int)(t >> 5);
    if (wave >= nTiles) return;
    int lane = threadIdx.x & 31;
    int half = (lane >> 4) & 1;
    int l15  = lane & 15;
    int row  = wave * 16 + l15;
    int koff = half * 8;
    int mbase = wave * 16 + half * 8;

    const v16bf* Wv = (const v16bf*)Wswz;
    v8f acc[8];
    #pragma unroll
    for (int nt = 0; nt < 8; ++nt) {
        #pragma unroll
        for (int g = 0; g < 8; ++g)
            acc[nt][g] = Out[(long)(mbase + g) * D + nt * 16 + l15];
    }

    const float* xrow = X + (long)row * D;
    #pragma unroll
    for (int kt = 0; kt < 4; ++kt) {
        int kb = kt * 32 + koff;
        v16bf av = make_afrag(*(const float4*)(xrow + kb),
                              *(const float4*)(xrow + kb + 4),
                              *(const float4*)(xrow + kb + 16),
                              *(const float4*)(xrow + kb + 20));
        #pragma unroll
        for (int nt = 0; nt < 8; ++nt) {
            v16bf bv = Wv[(kt * 8 + nt) * 32 + lane];
            acc[nt] = __builtin_amdgcn_wmma_f32_16x16x32_bf16(
                false, av, false, bv, (short)0, acc[nt], false, false);
        }
    }
    #pragma unroll
    for (int nt = 0; nt < 8; ++nt) {
        #pragma unroll
        for (int g = 0; g < 8; ++g)
            Out[(long)(mbase + g) * D + nt * 16 + l15] = acc[nt][g];
    }
}

// G1[g][:] = relu(concat(Xt[home[g]], Xt[away[g]]) @ W[256,128] + b)
__global__ void k_games_gemm(const float* __restrict__ Xt, const int* __restrict__ home,
                             const int* __restrict__ away, const bf16_t* __restrict__ Wswz,
                             const float* __restrict__ bvec, float* __restrict__ G1,
                             int nTiles) {
    long t = (long)blockIdx.x * blockDim.x + threadIdx.x;
    int wave = (int)(t >> 5);
    if (wave >= nTiles) return;
    int lane = threadIdx.x & 31;
    int half = (lane >> 4) & 1;
    int l15  = lane & 15;
    int g    = wave * 16 + l15;
    const float* hrow = Xt + (long)home[g] * D;
    const float* arow = Xt + (long)away[g] * D;
    int koff = half * 8;

    const v16bf* Wv = (const v16bf*)Wswz;
    v8f acc[8] = {};
    #pragma unroll
    for (int kt = 0; kt < 8; ++kt) {
        int kb = kt * 32 + koff;
        const float* srow = (kb < D) ? (hrow + kb) : (arow + (kb - D));
        v16bf av = make_afrag(*(const float4*)(srow),
                              *(const float4*)(srow + 4),
                              *(const float4*)(srow + 16),
                              *(const float4*)(srow + 20));
        #pragma unroll
        for (int nt = 0; nt < 8; ++nt) {
            v16bf bv = Wv[(kt * 8 + nt) * 32 + lane];
            acc[nt] = __builtin_amdgcn_wmma_f32_16x16x32_bf16(
                false, av, false, bv, (short)0, acc[nt], false, false);
        }
    }
    int mbase = wave * 16 + half * 8;
    #pragma unroll
    for (int nt = 0; nt < 8; ++nt) {
        int n = nt * 16 + l15;
        #pragma unroll
        for (int gg = 0; gg < 8; ++gg) {
            long idx = (long)(mbase + gg) * D + n;
            G1[idx] = fmaxf(acc[nt][gg] + bvec[n], 0.f);
        }
    }
}

// per-game 128->3 + log_softmax
__global__ void k_head2(const float* __restrict__ G1, const float* __restrict__ W,
                        const float* __restrict__ b, float* __restrict__ out, int N) {
    int g = blockIdx.x * blockDim.x + threadIdx.x;
    if (g >= N) return;
    float z0 = b[0], z1 = b[1], z2 = b[2];
    const float* row = G1 + (long)g * D;
    for (int k = 0; k < D; ++k) {
        float x = row[k];
        z0 += x * W[k * 3 + 0];
        z1 += x * W[k * 3 + 1];
        z2 += x * W[k * 3 + 2];
    }
    float mx  = fmaxf(z0, fmaxf(z1, z2));
    float lse = mx + logf(expf(z0 - mx) + expf(z1 - mx) + expf(z2 - mx));
    out[g * 3 + 0] = z0 - lse;
    out[g * 3 + 1] = z1 - lse;
    out[g * 3 + 2] = z2 - lse;
}

// ---------------- host orchestration ----------------

extern "C" void kernel_launch(void* const* d_in, const int* in_sizes, int n_in,
                              void* d_out, int out_size, void* d_ws, size_t ws_size,
                              hipStream_t stream) {
    (void)n_in; (void)out_size; (void)ws_size;
    auto F = [&](int i) { return (const float*)d_in[i]; };
    auto I = [&](int i) { return (const int*)d_in[i]; };

    // Input order (insertion order of setup_inputs + depth-first params flatten):
    // 0 x_team 1 x_player 2 e_win 3 e_loss 4 e_tie 5 e_playedin 6 e_used
    // 7 e_pbefore 8 e_pafter 9 e_tbefore 10 e_tafter 11 home 12 away 13 emb
    // per layer L (base 14 + 22*L):
    //   +0 win_W +1 win_b +2 loss_W +3 loss_b +4 tie_W +5 tie_b
    //   +6 pbefore_W +7 pbefore_b +8 pafter_W +9 pafter_b
    //   +10 tbefore_W +11 tbefore_b +12 tafter_W +13 tafter_b
    //   +14 playedin_W +15 playedin_as +16 playedin_ad +17 playedin_b
    //   +18 used_W +19 used_as +20 used_ad +21 used_b
    // 58 fc0_W 59 fc0_b 60 fc1_W 61 fc1_b

    const long SZ_T = (long)N_TEAM * D, SZ_P = (long)N_PLAYER * D;
    float* ws  = (float*)d_ws;
    float* xt0 = ws;
    float* xp0 = xt0 + SZ_T;
    float* xt1 = xp0 + SZ_P;
    float* xp1 = xt1 + SZ_T;
    float* Hb  = xp1 + SZ_P;                   // aggregation buffer, up to N_PLAYER x D
    float* G1  = Hb + SZ_P;                    // fc hidden, N_GAMES x D
    float* deg = G1 + (long)N_GAMES * D;       // N_PLAYER
    float* als = deg + N_PLAYER;
    float* ald = als + N_PLAYER;
    float* emx = ald + N_PLAYER;
    float* den = emx + N_PLAYER;
    float* wab = den + N_PLAYER;               // 256 floats
    bf16_t* Wswz = (bf16_t*)(wab + 256);       // up to 8*8*32*16 bf16 = 64 KB

    const int B = 256;
    auto blocks = [&](long n) { return (int)((n + B - 1) / B); };

    // embedding gather
    k_gather_emb<<<blocks((long)N_TEAM * 32), B, 0, stream>>>(F(13), I(0), xt0, N_TEAM);
    k_gather_emb<<<blocks((long)N_PLAYER * 32), B, 0, stream>>>(F(13), I(1), xp0, N_PLAYER);

    auto gcn = [&](const float* X, const int* ei, int E, const float* W, float* T, int N) {
        k_fill<<<blocks(N), B, 0, stream>>>(deg, N, 1.0f);
        k_edge_deg<<<blocks(E), B, 0, stream>>>(ei + E, E, deg);
        k_rsqrt<<<blocks(N), B, 0, stream>>>(deg, N);
        k_selfloop<<<blocks((long)N * D), B, 0, stream>>>(X, deg, Hb, N);
        k_gcn_scatter<<<blocks((long)E * 32), B, 0, stream>>>(X, ei, ei + E, E, deg, Hb);
        k_wswz<<<blocks(4 * 8 * 32), B, 0, stream>>>(W, Wswz, 4);
        int tiles = N / 16;
        k_gemm_acc<<<blocks((long)tiles * 32), B, 0, stream>>>(Hb, Wswz, T, tiles);
    };

    auto gat = [&](const float* Xs, const float* Xd, const int* ei, int E, const float* W,
                   const float* as_, const float* ad_, float* T, int Ns, int Nd) {
        k_wa<<<1, 128, 0, stream>>>(W, as_, wab);
        k_wa<<<1, 128, 0, stream>>>(W, ad_, wab + 128);
        k_dot_rows<<<blocks((long)Ns * 32), B, 0, stream>>>(Xs, wab, als, Ns);
        k_dot_rows<<<blocks((long)Nd * 32), B, 0, stream>>>(Xd, wab + 128, ald, Nd);
        k_fill<<<blocks(Nd), B, 0, stream>>>(emx, Nd, -1e30f);
        k_fill<<<blocks(Nd), B, 0, stream>>>(den, Nd, 0.f);
        int m = (Ns < Nd) ? Ns : Nd;
        int V = E + m;
        k_gat_max<<<blocks(V), B, 0, stream>>>(ei, ei + E, E, m, als, ald, emx);
        k_gat_den<<<blocks(V), B, 0, stream>>>(ei, ei + E, E, m, als, ald, emx, den);
        k_fill<<<blocks((long)Nd * D), B, 0, stream>>>(Hb, (long)Nd * D, 0.f);
        k_gat_scatter<<<blocks((long)V * 32), B, 0, stream>>>(Xs, ei, ei + E, E, m,
                                                              als, ald, emx, den, Hb);
        k_wswz<<<blocks(4 * 8 * 32), B, 0, stream>>>(W, Wswz, 4);
        int tiles = Nd / 16;
        k_gemm_acc<<<blocks((long)tiles * 32), B, 0, stream>>>(Hb, Wswz, T, tiles);
    };

    float* cxt = xt0; float* cxp = xp0;
    float* nxt = xt1; float* nxp = xp1;

    for (int L = 0; L < 2; ++L) {
        int pb = 14 + L * 22;
        // init accumulators with summed biases
        k_init_bias<<<blocks(SZ_T), B, 0, stream>>>(nxt, N_TEAM,
            F(pb + 1), F(pb + 3), F(pb + 5), F(pb + 11), F(pb + 13), F(pb + 17));
        k_init_bias<<<blocks(SZ_P), B, 0, stream>>>(nxp, N_PLAYER,
            F(pb + 21), F(pb + 7), F(pb + 9), nullptr, nullptr, nullptr);

        // team GCN relations: win, loss, tie, tbefore, tafter
        gcn(cxt, I(2),  in_sizes[2]  / 2, F(pb + 0),  nxt, N_TEAM);
        gcn(cxt, I(3),  in_sizes[3]  / 2, F(pb + 2),  nxt, N_TEAM);
        gcn(cxt, I(4),  in_sizes[4]  / 2, F(pb + 4),  nxt, N_TEAM);
        gcn(cxt, I(9),  in_sizes[9]  / 2, F(pb + 10), nxt, N_TEAM);
        gcn(cxt, I(10), in_sizes[10] / 2, F(pb + 12), nxt, N_TEAM);
        // GAT playedin: players -> teams
        gat(cxp, cxt, I(5), in_sizes[5] / 2, F(pb + 14), F(pb + 15), F(pb + 16),
            nxt, N_PLAYER, N_TEAM);
        // GAT used: teams -> players
        gat(cxt, cxp, I(6), in_sizes[6] / 2, F(pb + 18), F(pb + 19), F(pb + 20),
            nxp, N_TEAM, N_PLAYER);
        // player GCN relations: pbefore, pafter
        gcn(cxp, I(7), in_sizes[7] / 2, F(pb + 6), nxp, N_PLAYER);
        gcn(cxp, I(8), in_sizes[8] / 2, F(pb + 8), nxp, N_PLAYER);

        if (L == 0) {
            k_relu<<<blocks(SZ_T), B, 0, stream>>>(nxt, SZ_T);
            k_relu<<<blocks(SZ_P), B, 0, stream>>>(nxp, SZ_P);
        }
        float* tmp;
        tmp = cxt; cxt = nxt; nxt = tmp;
        tmp = cxp; cxp = nxp; nxp = tmp;
    }

    // head: concat-gather GEMM (K=256) + 128->3 + log_softmax
    k_wswz<<<blocks(8 * 8 * 32), B, 0, stream>>>(F(58), Wswz, 8);
    int gtiles = N_GAMES / 16;
    k_games_gemm<<<blocks((long)gtiles * 32), B, 0, stream>>>(
        cxt, I(11), I(12), Wswz, F(59), G1, gtiles);
    k_head2<<<blocks(N_GAMES), 128, 0, stream>>>(G1, F(60), F(61), (float*)d_out, N_GAMES);
}
